// LinkPredictor_38096359916184
// MI455X (gfx1250) — matrix operand order repaired and checked by reading, not compiled, not run on hardware
//
#include <hip/hip_runtime.h>

typedef __attribute__((ext_vector_type(16))) _Float16 v16h;
typedef __attribute__((ext_vector_type(8)))  _Float16 v8h;
typedef __attribute__((ext_vector_type(8)))  float    v8f;

#define NNODES 50000
#define DFEAT  128
#define NEDGES 800000
#define NEVAL  100000

// ---------------- degree / normalization ----------------
__global__ void k_init_deg(float* deg, int n) {
    int i = blockIdx.x * blockDim.x + threadIdx.x;
    if (i < n) deg[i] = 1.0f;               // self-loop contributes +1
}

__global__ void k_deg_scatter(const int* __restrict__ dst, float* __restrict__ deg, int ne) {
    int e = blockIdx.x * blockDim.x + threadIdx.x;
    if (e < ne) atomicAdd(&deg[dst[e]], 1.0f);
}

__global__ void k_rsqrt_inplace(float* d, int n) {
    int i = blockIdx.x * blockDim.x + threadIdx.x;
    if (i < n) d[i] = rsqrtf(d[i]);
}

// ---------------- fp32 -> f16 (optional ReLU) ----------------
__global__ void k_f32_to_f16(const float* __restrict__ in, _Float16* __restrict__ out,
                             int n, int do_relu) {
    int i = blockIdx.x * blockDim.x + threadIdx.x;
    if (i < n) {
        float v = in[i];
        if (do_relu) v = v > 0.0f ? v : 0.0f;
        out[i] = (_Float16)v;
    }
}

// ---------------- WMMA GEMM: C[M x 128] = A[M x 128] * B[128 x 128] ----------------
// One 16x16 C tile per wave; block = 8 waves covering the full N=128 strip of one
// 16-row M tile. K consumed in 4 chunks of 32 via v_wmma_f32_16x16x32_f16.
__global__ void __launch_bounds__(256) k_gemm_f16(const _Float16* __restrict__ A,
                                                  const _Float16* __restrict__ B,
                                                  float* __restrict__ C, int M) {
    int wave = threadIdx.x >> 5;
    int lane = threadIdx.x & 31;
    int hl   = lane >> 4;               // lane group: 0 -> K base 0, 1 -> K base 8
    int m0   = blockIdx.x * 16;
    if (m0 >= M) return;
    int n0   = wave * 16;

    const _Float16* arow = A + (size_t)(m0 + (lane & 15)) * DFEAT;

    v8f acc = {};
#pragma unroll
    for (int kt = 0; kt < DFEAT; kt += 32) {
        // A 16x32 f16 tile: lane holds M=lane%16, halves e<8 -> K=kt+hl*8+e,
        // e>=8 -> K=kt+16+hl*8+(e-8): two contiguous 16B loads.
        v8h alo = *(const v8h*)(arow + kt + hl * 8);
        v8h ahi = *(const v8h*)(arow + kt + 16 + hl * 8);
        v16h a  = __builtin_shufflevector(alo, ahi,
                    0,1,2,3,4,5,6,7,8,9,10,11,12,13,14,15);
        // B 32x16 f16 tile: lane holds N=n0+lane%16, element e -> K same as A (table order).
        v16h b;
#pragma unroll
        for (int e = 0; e < 16; ++e)
            b[e] = B[(size_t)(kt + (e < 8 ? hl * 8 + e : 16 + hl * 8 + (e - 8))) * DFEAT + n0 + (lane & 15)];
        acc = __builtin_amdgcn_wmma_f32_16x16x32_f16(
                  /*neg_a=*/false, a, /*neg_b=*/false, b,
                  /*c_mod=*/(short)0, acc, /*reuse_a=*/false, /*reuse_b=*/false);
    }
    // C 16x16 f32: lane -> N = lane&15, VGPR r -> M = r + 8*hl
    float* crow = C + (size_t)(m0 + hl * 8) * DFEAT + n0 + (lane & 15);
#pragma unroll
    for (int r = 0; r < 8; ++r) crow[(size_t)r * DFEAT] = acc[r];
}

// ---------------- self-loop + bias init: out = h * dinv^2 + b ----------------
__global__ void k_self_init(const float* __restrict__ h, const float* __restrict__ dinv,
                            const float* __restrict__ bias, float* __restrict__ out,
                            int total) {
    int i = blockIdx.x * blockDim.x + threadIdx.x;
    if (i < total) {
        int node = i >> 7;              // DFEAT == 128
        int c    = i & 127;
        float di = dinv[node];
        out[i] = h[i] * di * di + bias[c];
    }
}

// ---------------- edge scatter: out[dst] += h[src] * dinv[src]*dinv[dst] ----------------
// One edge per 128-thread block; each wave issues 32 consecutive-column f32 atomics
// (coalesced within cachelines, resolved in L2 — h fits in the 192MB L2).
__global__ void __launch_bounds__(128) k_edge_scatter(const int* __restrict__ src,
                                                      const int* __restrict__ dst,
                                                      const float* __restrict__ h,
                                                      const float* __restrict__ dinv,
                                                      float* __restrict__ out, int ne) {
    int e = blockIdx.x;
    if (e >= ne) return;
    int s = src[e], d = dst[e];
    float norm = dinv[s] * dinv[d];
    int c = threadIdx.x;
    atomicAdd(&out[(size_t)d * DFEAT + c], h[(size_t)s * DFEAT + c] * norm);
}

// ---------------- link scores: one wave32 per dot product ----------------
__global__ void __launch_bounds__(256) k_scores(const float* __restrict__ z,
                                                const int* __restrict__ pidx,
                                                const int* __restrict__ nidx,
                                                float* __restrict__ out) {
    int gw   = blockIdx.x * 8 + (threadIdx.x >> 5);
    int lane = threadIdx.x & 31;
    if (gw >= 2 * NEVAL) return;
    const int* a; const int* b; float* o; int i;
    if (gw < NEVAL) { i = gw;         a = pidx; b = pidx + NEVAL; o = out; }
    else            { i = gw - NEVAL; a = nidx; b = nidx + NEVAL; o = out + NEVAL; }
    const float* zu = z + (size_t)a[i] * DFEAT;
    const float* zv = z + (size_t)b[i] * DFEAT;
    float s = 0.0f;
#pragma unroll
    for (int t = 0; t < 4; ++t) s += zu[lane + 32 * t] * zv[lane + 32 * t];
#pragma unroll
    for (int m = 16; m >= 1; m >>= 1) s += __shfl_xor(s, m, 32);
    if (lane == 0) o[i] = s;
}

extern "C" void kernel_launch(void* const* d_in, const int* in_sizes, int n_in,
                              void* d_out, int out_size, void* d_ws, size_t ws_size,
                              hipStream_t stream) {
    const float* x   = (const float*)d_in[0];
    const float* W1  = (const float*)d_in[1];
    const float* b1  = (const float*)d_in[2];
    const float* W2  = (const float*)d_in[3];
    const float* b2  = (const float*)d_in[4];
    const int*   eix = (const int*)d_in[5];
    const int*   pix = (const int*)d_in[6];
    const int*   nix = (const int*)d_in[7];
    const int* esrc = eix;
    const int* edst = eix + NEDGES;

    // workspace carve-up (256B-aligned chunks)
    char* ws = (char*)d_ws;
    size_t off = 0;
    auto carve = [&](size_t bytes) -> void* {
        void* p = ws + off;
        off += (bytes + 255) & ~(size_t)255;
        return p;
    };
    float*    dinv = (float*)   carve((size_t)NNODES * 4);
    _Float16* W1h  = (_Float16*)carve((size_t)DFEAT * DFEAT * 2);
    _Float16* W2h  = (_Float16*)carve((size_t)DFEAT * DFEAT * 2);
    _Float16* actH = (_Float16*)carve((size_t)NNODES * DFEAT * 2); // x_f16, reused as relu(z1)_f16
    float*    h    = (float*)   carve((size_t)NNODES * DFEAT * 4); // GEMM out, layer1 then layer2
    float*    z1   = (float*)   carve((size_t)NNODES * DFEAT * 4);
    float*    z2   = (float*)   carve((size_t)NNODES * DFEAT * 4);
    (void)ws_size; (void)in_sizes; (void)n_in; (void)out_size;

    const int totalF = NNODES * DFEAT;   // 6,400,000

    // normalization: deg = 1 + indegree ; dinv = rsqrt(deg)
    k_init_deg   <<<(NNODES + 255) / 256, 256, 0, stream>>>(dinv, NNODES);
    k_deg_scatter<<<(NEDGES + 255) / 256, 256, 0, stream>>>(edst, dinv, NEDGES);
    k_rsqrt_inplace<<<(NNODES + 255) / 256, 256, 0, stream>>>(dinv, NNODES);

    // f16 operands
    k_f32_to_f16<<<(totalF + 255) / 256, 256, 0, stream>>>(x,  actH, totalF, 0);
    k_f32_to_f16<<<(DFEAT * DFEAT + 255) / 256, 256, 0, stream>>>(W1, W1h, DFEAT * DFEAT, 0);
    k_f32_to_f16<<<(DFEAT * DFEAT + 255) / 256, 256, 0, stream>>>(W2, W2h, DFEAT * DFEAT, 0);

    // layer 1: h = x@W1 ; z1 = scatter(h) + h*dinv^2 + b1
    k_gemm_f16    <<<NNODES / 16, 256, 0, stream>>>(actH, W1h, h, NNODES);
    k_self_init   <<<(totalF + 255) / 256, 256, 0, stream>>>(h, dinv, b1, z1, totalF);
    k_edge_scatter<<<NEDGES, 128, 0, stream>>>(esrc, edst, h, dinv, z1, NEDGES);

    // layer 2: h = relu(z1)@W2 ; z2 = scatter(h) + h*dinv^2 + b2
    k_f32_to_f16  <<<(totalF + 255) / 256, 256, 0, stream>>>(z1, actH, totalF, 1);
    k_gemm_f16    <<<NNODES / 16, 256, 0, stream>>>(actH, W2h, h, NNODES);
    k_self_init   <<<(totalF + 255) / 256, 256, 0, stream>>>(h, dinv, b2, z2, totalF);
    k_edge_scatter<<<NEDGES, 128, 0, stream>>>(esrc, edst, h, dinv, z2, NEDGES);

    // scores: 200k wave-level dot products
    float* out = (float*)d_out;
    k_scores<<<(2 * NEVAL) / 8, 256, 0, stream>>>(z2, pix, nix, out);
}
